// VQCodebook_55903294325259
// MI455X (gfx1250) — matrix operand order repaired and checked by reading, not compiled, not run on hardware
//
#include <hip/hip_runtime.h>
#include <hip/hip_bf16.h>

typedef __bf16 v16bf __attribute__((ext_vector_type(16)));
typedef float  v8f   __attribute__((ext_vector_type(8)));

#define NUM_CODES 8192
#define CODE_DIM  256
#define NTOK      65536

// Use CDNA5 async global->LDS copies for B staging (ASYNCcnt-tracked).
#define USE_ASYNC 1

// ---------------------------------------------------------------------------
// Kernel 1: convert codebook to bf16 + compute ||e_k||^2
// ---------------------------------------------------------------------------
__global__ __launch_bounds__(256) void vq_prep(const float* __restrict__ w,
                                               __bf16* __restrict__ wbf,
                                               float* __restrict__ wnorm) {
    const int k = blockIdx.x;
    const int d = threadIdx.x;
    float v = w[(size_t)k * CODE_DIM + d];
    wbf[(size_t)k * CODE_DIM + d] = (__bf16)v;
    float s = v * v;
    #pragma unroll
    for (int off = 16; off > 0; off >>= 1) s += __shfl_down(s, off, 32);
    __shared__ float red[8];
    const int wave = threadIdx.x >> 5, lane = threadIdx.x & 31;
    if (lane == 0) red[wave] = s;
    __syncthreads();
    if (threadIdx.x == 0) {
        float t = 0.f;
        #pragma unroll
        for (int i = 0; i < 8; ++i) t += red[i];
        wnorm[k] = t;
    }
}

// ---------------------------------------------------------------------------
// Kernel 2: fused GEMM + argmin.
// 256 threads = 8 waves. Wave tile: M=32 rows (two register-resident bf16
// A-fragment sets, 128 VGPRs) x N=32 codes per chunk (4 accumulator chains;
// each B fragment feeds 2 WMMAs). B chunks (32 codes x 256 depth bf16, 16 KB)
// double-buffered in LDS, staged with global_load_async_to_lds_b128 while the
// previous chunk is computed; one barrier per chunk.
// score(n,k) = ||e_k||^2 - 2 z_n.e_k   (||z||^2 constant per row -> dropped)
// ---------------------------------------------------------------------------
__global__ __launch_bounds__(256) void vq_argmin(const float* __restrict__ z,
                                                 const __bf16* __restrict__ wbf,
                                                 const float* __restrict__ wnorm,
                                                 int* __restrict__ idx_out,
                                                 float* __restrict__ idx_out_f) {
    __shared__ uint4 ldsB[2][1024];  // 2 x (32 codes * 256 depth * 2B) = 32 KB

    const int tid  = threadIdx.x;
    const int wave = tid >> 5;
    const int lane = tid & 31;
    const int half = lane >> 4;   // 0: lanes 0-15, 1: lanes 16-31
    const int l16  = lane & 15;
    const int row0 = blockIdx.x * 256 + wave * 32;  // wave's 32-row tile base

    // ---- Load A fragments (two 16-row tiles), contiguous K layout:
    //   half 0: K in {0..15};  half 1: K in {16..31}
    v16bf afrag0[8], afrag1[8];
    {
        const float* zr0 = z + (size_t)(row0 + l16) * CODE_DIM;
        const float* zr1 = z + (size_t)(row0 + 16 + l16) * CODE_DIM;
        #pragma unroll
        for (int t = 0; t < 8; ++t) {
            const int base = 32 * t + 16 * half;
            #pragma unroll
            for (int i = 0; i < 8; ++i) {
                afrag0[t][i]     = (__bf16)zr0[base + i];
                afrag0[t][8 + i] = (__bf16)zr0[base + 8 + i];
                afrag1[t][i]     = (__bf16)zr1[base + i];
                afrag1[t][8 + i] = (__bf16)zr1[base + 8 + i];
            }
        }
    }

    // Per-lane running argmin: j = mt*8 + r tracks row row0 + 16*mt + 8*half + r
    float minv[16];
    int   mini[16];
    #pragma unroll
    for (int j = 0; j < 16; ++j) { minv[j] = 3.4e38f; mini[j] = 0; }

    const int NCHUNK = NUM_CODES / 32;  // 256 chunks of 32 codes

    // ---- Prologue: stage chunk 0 into buffer 0
    {
        const uint4* src = (const uint4*)wbf;  // chunk 0 at offset 0
#if USE_ASYNC
        #pragma unroll
        for (int j = 0; j < 4; ++j) {
            unsigned lo = (unsigned)(uintptr_t)(&ldsB[0][tid + 256 * j]);
            asm volatile("global_load_async_to_lds_b128 %0, %1, off"
                         :: "v"(lo), "v"(src + tid + 256 * j) : "memory");
        }
        asm volatile("s_wait_asynccnt 0x0" ::: "memory");
#else
        #pragma unroll
        for (int j = 0; j < 4; ++j) ldsB[0][tid + 256 * j] = src[tid + 256 * j];
#endif
        __syncthreads();
    }

    for (int ci = 0; ci < NCHUNK; ++ci) {
        const int c0 = ci * 32;
        const int p  = ci & 1;

        // ---- Stage next chunk into the other buffer (overlaps compute)
#if USE_ASYNC
        if (ci + 1 < NCHUNK) {
            const uint4* src = (const uint4*)(wbf + (size_t)(c0 + 32) * CODE_DIM);
            #pragma unroll
            for (int j = 0; j < 4; ++j) {
                unsigned lo = (unsigned)(uintptr_t)(&ldsB[1 - p][tid + 256 * j]);
                asm volatile("global_load_async_to_lds_b128 %0, %1, off"
                             :: "v"(lo), "v"(src + tid + 256 * j) : "memory");
            }
        }
#else
        uint4 streg[4];
        if (ci + 1 < NCHUNK) {
            const uint4* src = (const uint4*)(wbf + (size_t)(c0 + 32) * CODE_DIM);
            #pragma unroll
            for (int j = 0; j < 4; ++j) streg[j] = src[tid + 256 * j];
        }
#endif

        const float wn0 = wnorm[c0 + l16];
        const float wn1 = wnorm[c0 + 16 + l16];

        // ---- Compute: 4 accumulator chains (2 M-tiles x 2 code-fragments)
        v8f acc00 = {}, acc01 = {}, acc10 = {}, acc11 = {};
        #pragma unroll
        for (int t = 0; t < 8; ++t) {
            // B layout (32x16 bf16): column N = l16, half selects K 16h..16h+15
            const int dofs = 32 * t + 16 * half;  // depth offset, bf16 units
            union { uint4 u[2]; v16bf v; } b0, b1;
            const int u0 = l16 * 32 + (dofs >> 3);
            b0.u[0] = ldsB[p][u0];
            b0.u[1] = ldsB[p][u0 + 1];
            const int u1 = (16 + l16) * 32 + (dofs >> 3);
            b1.u[0] = ldsB[p][u1];
            b1.u[1] = ldsB[p][u1 + 1];
            acc00 = __builtin_amdgcn_wmma_f32_16x16x32_bf16(
                false, afrag0[t], false, b0.v, (short)0, acc00, false, false);
            acc10 = __builtin_amdgcn_wmma_f32_16x16x32_bf16(
                false, afrag1[t], false, b0.v, (short)0, acc10, false, false);
            acc01 = __builtin_amdgcn_wmma_f32_16x16x32_bf16(
                false, afrag0[t], false, b1.v, (short)0, acc01, false, false);
            acc11 = __builtin_amdgcn_wmma_f32_16x16x32_bf16(
                false, afrag1[t], false, b1.v, (short)0, acc11, false, false);
        }

        // ---- Fold scores into running argmin
        // C layout: lane holds N = l16, VGPR r -> M = 8*half + r (per tile)
        #pragma unroll
        for (int r = 0; r < 8; ++r) {
            float s;
            s = wn0 - 2.0f * acc00[r];
            if (s < minv[r])     { minv[r] = s;     mini[r] = c0 + l16; }
            s = wn1 - 2.0f * acc01[r];
            if (s < minv[r])     { minv[r] = s;     mini[r] = c0 + 16 + l16; }
            s = wn0 - 2.0f * acc10[r];
            if (s < minv[8 + r]) { minv[8 + r] = s; mini[8 + r] = c0 + l16; }
            s = wn1 - 2.0f * acc11[r];
            if (s < minv[8 + r]) { minv[8 + r] = s; mini[8 + r] = c0 + 16 + l16; }
        }

        // ---- Close the pipeline stage
#if USE_ASYNC
        asm volatile("s_wait_asynccnt 0x0" ::: "memory");
#else
        if (ci + 1 < NCHUNK) {
            #pragma unroll
            for (int j = 0; j < 4; ++j) ldsB[1 - p][tid + 256 * j] = streg[j];
        }
#endif
        __syncthreads();
    }

    // ---- Cross-lane argmin within each 16-lane group (same rows, diff codes)
    #pragma unroll
    for (int j = 0; j < 16; ++j) {
        float v = minv[j];
        int   i = mini[j];
        #pragma unroll
        for (int off = 8; off >= 1; off >>= 1) {
            const float ov = __shfl_xor(v, off, 32);
            const int   oi = __shfl_xor(i, off, 32);
            if (ov < v || (ov == v && oi < i)) { v = ov; i = oi; }
        }
        minv[j] = v; mini[j] = i;
    }
    if (l16 == 0) {
        #pragma unroll
        for (int mt = 0; mt < 2; ++mt) {
            #pragma unroll
            for (int r = 0; r < 8; ++r) {
                const int row = row0 + 16 * mt + 8 * half + r;
                idx_out[row]   = mini[mt * 8 + r];
                idx_out_f[row] = (float)mini[mt * 8 + r];
            }
        }
    }
}

// ---------------------------------------------------------------------------
// Kernel 3: gather z_q = weight[idx] (fp32, exact) + partial sum of (z_q-z)^2
// ---------------------------------------------------------------------------
__global__ __launch_bounds__(256) void vq_gather(const float* __restrict__ z,
                                                 const float* __restrict__ w,
                                                 const int* __restrict__ idx,
                                                 float* __restrict__ zq,
                                                 float* __restrict__ partial) {
    const int tid  = threadIdx.x;
    const int tok0 = blockIdx.x * 32;
    float acc = 0.f;
    for (int t = 0; t < 32; ++t) {
        const int n = tok0 + t;
        const int k = idx[n];
        const float wv = w[(size_t)k * CODE_DIM + tid];
        const float zv = z[(size_t)n * CODE_DIM + tid];
        zq[(size_t)n * CODE_DIM + tid] = wv;
        const float d = wv - zv;
        acc += d * d;
    }
    #pragma unroll
    for (int off = 16; off > 0; off >>= 1) acc += __shfl_down(acc, off, 32);
    __shared__ float red[8];
    const int wave = tid >> 5, lane = tid & 31;
    if (lane == 0) red[wave] = acc;
    __syncthreads();
    if (tid == 0) {
        float t = 0.f;
        #pragma unroll
        for (int i = 0; i < 8; ++i) t += red[i];
        partial[blockIdx.x] = t;
    }
}

// ---------------------------------------------------------------------------
// Kernel 4: deterministic final reduction -> both loss scalars
// ---------------------------------------------------------------------------
__global__ __launch_bounds__(256) void vq_loss(const float* __restrict__ partial,
                                               float* __restrict__ out_losses) {
    float acc = 0.f;
    for (int i = threadIdx.x; i < NTOK / 32; i += 256) acc += partial[i];
    #pragma unroll
    for (int off = 16; off > 0; off >>= 1) acc += __shfl_down(acc, off, 32);
    __shared__ float red[8];
    const int wave = threadIdx.x >> 5, lane = threadIdx.x & 31;
    if (lane == 0) red[wave] = acc;
    __syncthreads();
    if (threadIdx.x == 0) {
        float t = 0.f;
        #pragma unroll
        for (int i = 0; i < 8; ++i) t += red[i];
        const float m = t * (1.0f / (float)((size_t)NTOK * CODE_DIM));
        out_losses[0] = m;  // codebook_loss
        out_losses[1] = m;  // commitment_loss (identical forward value)
    }
}

// ---------------------------------------------------------------------------
extern "C" void kernel_launch(void* const* d_in, const int* in_sizes, int n_in,
                              void* d_out, int out_size, void* d_ws, size_t ws_size,
                              hipStream_t stream) {
    const float* z = (const float*)d_in[0];   // [16,64,64,256] fp32
    const float* w = (const float*)d_in[1];   // [8192,256] fp32

    float* out    = (float*)d_out;
    float* zq     = out;                                   // 16,777,216
    float* losses = out + (size_t)NTOK * CODE_DIM;         // 2
    float* idxF   = losses + 2;                            // 65,536

    char* ws = (char*)d_ws;
    __bf16* wbf    = (__bf16*)ws;                                          // 4 MB
    float*  wnorm  = (float*)(ws + (size_t)NUM_CODES * CODE_DIM * 2);      // 32 KB
    int*    idxI   = (int*)((char*)wnorm + (size_t)NUM_CODES * 4);         // 256 KB
    float*  partial = (float*)((char*)idxI + (size_t)NTOK * 4);            // 8 KB

    vq_prep  <<<NUM_CODES, 256, 0, stream>>>(w, wbf, wnorm);
    vq_argmin<<<NTOK / 256, 256, 0, stream>>>(z, wbf, wnorm, idxI, idxF);
    vq_gather<<<NTOK / 32, 256, 0, stream>>>(z, w, idxI, zq, partial);
    vq_loss  <<<1, 256, 0, stream>>>(partial, losses);
}
